// PerformerAEPositionalEncoding_42399917146171
// MI455X (gfx1250) — matrix operand order repaired and checked by reading, not compiled, not run on hardware
//
#include <hip/hip_runtime.h>

typedef unsigned short ushort_t;
typedef __attribute__((ext_vector_type(16))) __bf16 bf16x16;
typedef __attribute__((ext_vector_type(8)))  float  f32x8;

#define T_TOK   65536      // B*W
#define SEQ_W   8192
#define DDIM    128
#define HEADS   8
#define DH      16
#define MFEAT   44
#define MPAD    48
#define DFF     512
#define LATDIM  8
#define NLAYERS 8
#define RATIO   0.15075567228888181f   // 44^-0.5
#define GELU_K  0.7071067811865476f

__device__ __forceinline__ ushort_t f2bf(float f) {
  unsigned u = __float_as_uint(f);
  u = (u + 0x7FFFu + ((u >> 16) & 1u)) >> 16;
  return (ushort_t)u;
}
__device__ __forceinline__ unsigned f2ord(float f) {
  unsigned u = __float_as_uint(f);
  return (u & 0x80000000u) ? ~u : (u | 0x80000000u);
}
__device__ __forceinline__ float ord2f(unsigned k) {
  unsigned u = (k & 0x80000000u) ? (k & 0x7FFFFFFFu) : ~k;
  return __uint_as_float(u);
}
__device__ __forceinline__ float gelu_exact(float x) {
  return 0.5f * x * (1.0f + erff(x * GELU_K));
}

// ---------------- weight repack: fp32 [K,N] -> bf16 [N,K] (transposed) -------
__global__ void pack_wt(const float* __restrict__ src, ushort_t* __restrict__ dst,
                        int K, int N) {
  int idx = blockIdx.x * blockDim.x + threadIdx.x;
  if (idx >= K * N) return;
  int n = idx / K, k = idx - n * K;
  dst[idx] = f2bf(src[(size_t)k * N + n]);
}

// ---------------- elementwise fp32 -> bf16 ----------------------------------
__global__ void pack_bf16(const float* __restrict__ src, ushort_t* __restrict__ dst,
                          int n) {
  int idx = blockIdx.x * blockDim.x + threadIdx.x;
  if (idx < n) dst[idx] = f2bf(src[idx]);
}

// ---------------- WMMA GEMM: C[T x N] = A[T x K](bf16) * Bt[N x K](bf16) -----
// epilogues: 0 = bias -> f32 ; 1 = bias + fixed pos-emb -> f32 ;
//            2 = bias + residual -> f32 ; 3 = bias + gelu -> bf16
union AFrag { bf16x16 v; uint4 q[2]; };

template <int EPI>
__global__ __launch_bounds__(256)
void gemm_bf16_wmma(const ushort_t* __restrict__ A, const ushort_t* __restrict__ Bt,
                    const float* __restrict__ bias, const float* __restrict__ resid,
                    float* __restrict__ Cf, ushort_t* __restrict__ Cb,
                    int N, int K) {
  const int lane = threadIdx.x & 31;
  const int wave = threadIdx.x >> 5;       // 0..7
  const int hi   = lane >> 4;              // 0/1
  const int mr   = lane & 15;
  const int wm   = wave & 3;               // 4 row-tiles of 32
  const int wn   = wave >> 2;              // 2 col-tiles of 32
  const int row0 = blockIdx.y * 128 + wm * 32;
  const int col0 = blockIdx.x * 64  + wn * 32;

  f32x8 c[2][2] = {};
  for (int k0 = 0; k0 < K; k0 += 32) {
    AFrag a[2], b[2];
#pragma unroll
    for (int i = 0; i < 2; ++i) {
      // A fragment: row = row0+i*16+mr, K chunks [hi*8, hi*8+8) and +16
      const ushort_t* pa = A + (size_t)(row0 + i * 16 + mr) * K + k0 + hi * 8;
      a[i].q[0] = *(const uint4*)(pa);
      a[i].q[1] = *(const uint4*)(pa + 16);
      // B fragment: col n = col0+i*16+mr, K chunks [hi*8, hi*8+8) and +16
      const ushort_t* pb = Bt + (size_t)(col0 + i * 16 + mr) * K + k0 + hi * 8;
      b[i].q[0] = *(const uint4*)(pb);
      b[i].q[1] = *(const uint4*)(pb + 16);
      if (k0 + 32 < K) {                       // gfx1250 global_prefetch_b8
        __builtin_prefetch(pa + 32, 0, 3);
        __builtin_prefetch(pb + 32, 0, 3);
      }
    }
#pragma unroll
    for (int i = 0; i < 2; ++i)
#pragma unroll
      for (int j = 0; j < 2; ++j)
        c[i][j] = __builtin_amdgcn_wmma_f32_16x16x32_bf16(
            false, a[i].v, false, b[j].v, (short)0, c[i][j], false, false);
  }

#pragma unroll
  for (int i = 0; i < 2; ++i) {
#pragma unroll
    for (int j = 0; j < 2; ++j) {
#pragma unroll
      for (int r = 0; r < 8; ++r) {
        int row = row0 + i * 16 + hi * 8 + r;
        int col = col0 + j * 16 + mr;
        float acc = c[i][j][r] + bias[col];
        if (EPI == 1) {  // fixed positional embedding [sin | cos]
          int pos = row & (SEQ_W - 1);
          int cc  = col & 63;
          float ang = (float)pos * __powf(10000.f, -(float)cc / 64.f);
          acc += (col < 64) ? __sinf(ang) : __cosf(ang);
        }
        if (EPI == 2) acc += resid[(size_t)row * N + col];
        size_t o = (size_t)row * N + col;
        if (EPI == 3) Cb[o] = f2bf(gelu_exact(acc));
        else          Cf[o] = acc;
      }
    }
  }
}

// ---------------- LayerNorm: f32 in -> bf16 out (one wave per row) -----------
__global__ __launch_bounds__(256)
void layernorm_bf16(const float* __restrict__ x, const float* __restrict__ g,
                    const float* __restrict__ b, ushort_t* __restrict__ out) {
  int lane = threadIdx.x & 31;
  int row  = blockIdx.x * 8 + (threadIdx.x >> 5);
  const float* xr = x + (size_t)row * DDIM;
  float v[4], s = 0.f, s2 = 0.f;
#pragma unroll
  for (int i = 0; i < 4; ++i) {
    v[i] = xr[lane + 32 * i];
    s += v[i]; s2 += v[i] * v[i];
  }
  for (int m = 16; m > 0; m >>= 1) { s += __shfl_xor(s, m, 32); s2 += __shfl_xor(s2, m, 32); }
  float mu = s * (1.f / DDIM);
  float rs = rsqrtf(s2 * (1.f / DDIM) - mu * mu + 1e-5f);
#pragma unroll
  for (int i = 0; i < 4; ++i) {
    int c = lane + 32 * i;
    out[(size_t)row * DDIM + c] = f2bf((v[i] - mu) * rs * g[c] + b[c]);
  }
}

// ---------------- rotary helper -------------------------------------------
__device__ __forceinline__ void rotary16(const float* __restrict__ src, int pos,
                                         float* __restrict__ d) {
#pragma unroll
  for (int i = 0; i < 8; ++i) {
    float ang = (float)pos * __powf(10000.f, -(float)i / 8.f);
    float s = __sinf(ang), c = __cosf(ang);
    float e = src[2 * i], o = src[2 * i + 1];
    d[2 * i]     = e * c - o * s;
    d[2 * i + 1] = o * c + e * s;
  }
}

// ---------------- FAVOR+ query features ------------------------------------
__global__ __launch_bounds__(256)
void feat_q(const float* __restrict__ q, const float* __restrict__ proj,
            float* __restrict__ qp) {
  __shared__ float P[MFEAT * DH];
  for (int i = threadIdx.x; i < MFEAT * DH; i += 256) P[i] = proj[i];
  __syncthreads();
  int idx = blockIdx.x * 256 + threadIdx.x;      // token*8 + head
  int tok = idx >> 3, h = idx & 7;
  float d[DH];
  rotary16(q + (size_t)tok * DDIM + h * DH, tok & (SEQ_W - 1), d);
  float sq = 0.f;
#pragma unroll
  for (int i = 0; i < DH; ++i) sq += d[i] * d[i];
  float diag = 0.125f * sq;                      // 0.5 * dn^2, dn = 16^-0.25 = 0.5
  float dd[MFEAT], mx = -1e30f;
  for (int m = 0; m < MFEAT; ++m) {
    float a = 0.f;
#pragma unroll
    for (int i = 0; i < DH; ++i) a += d[i] * P[m * DH + i];
    a *= 0.5f;
    dd[m] = a; mx = fmaxf(mx, a);
  }
  float* dst = qp + (size_t)idx * MPAD;
  for (int m = 0; m < MFEAT; ++m) dst[m] = RATIO * (__expf(dd[m] - diag - mx) + 1e-4f);
}

// ---------------- FAVOR+ key features, phase 1: dd/diag + global max ---------
__global__ __launch_bounds__(256)
void feat_k1(const float* __restrict__ k, const float* __restrict__ proj,
             float* __restrict__ kp, unsigned* __restrict__ kmax) {
  __shared__ float P[MFEAT * DH];
  __shared__ unsigned smax[HEADS];
  if (threadIdx.x < HEADS) smax[threadIdx.x] = 0u;   // below any ordered-float key
  for (int i = threadIdx.x; i < MFEAT * DH; i += 256) P[i] = proj[i];
  __syncthreads();
  int idx = blockIdx.x * 256 + threadIdx.x;
  int tok = idx >> 3, h = idx & 7;
  float d[DH];
  rotary16(k + (size_t)tok * DDIM + h * DH, tok & (SEQ_W - 1), d);
  float sq = 0.f;
#pragma unroll
  for (int i = 0; i < DH; ++i) sq += d[i] * d[i];
  float diag = 0.125f * sq;
  float mx = -1e30f;
  float* dst = kp + (size_t)idx * MPAD;
  for (int m = 0; m < MFEAT; ++m) {
    float a = 0.f;
#pragma unroll
    for (int i = 0; i < DH; ++i) a += d[i] * P[m * DH + i];
    a *= 0.5f;
    dst[m] = a; mx = fmaxf(mx, a);
  }
  dst[MFEAT] = diag;
  atomicMax(&smax[h], f2ord(mx));
  __syncthreads();
  if (threadIdx.x < HEADS) {
    int b = blockIdx.x >> 8;                 // 256 blocks per batch element
    atomicMax(&kmax[b * HEADS + threadIdx.x], smax[threadIdx.x]);
  }
}

// ---------------- FAVOR+ key features, phase 2 ------------------------------
__global__ __launch_bounds__(256)
void feat_k2(float* __restrict__ kp, const unsigned* __restrict__ kmax) {
  int idx = blockIdx.x * 256 + threadIdx.x;
  int b = idx >> 16, h = idx & 7;
  float mx = ord2f(kmax[b * HEADS + h]);
  float* row = kp + (size_t)idx * MPAD;
  float diag = row[MFEAT];
  for (int m = 0; m < MFEAT; ++m)
    row[m] = RATIO * (__expf(row[m] - diag - mx) + 1e-4f);
}

__global__ void kmax_init(unsigned* kmax) { kmax[threadIdx.x] = 0u; }

// ---------------- ctx: per (b,h) [44 x 17] = kp^T @ [v | 1] ------------------
__global__ __launch_bounds__(256)
void ctx_kernel(const float* __restrict__ kp, const float* __restrict__ v,
                float* __restrict__ ctx) {
  int bh = blockIdx.x, b = bh >> 3, h = bh & 7;
  __shared__ float KP[32 * MPAD];
  __shared__ float VV[32 * DH];
  float acc[3] = {0.f, 0.f, 0.f};
  for (int n0 = 0; n0 < SEQ_W; n0 += 32) {
    __syncthreads();
    for (int i = threadIdx.x; i < 32 * MPAD; i += 256) {
      int t = i / MPAD, m = i - t * MPAD;
      KP[i] = kp[((size_t)((b * SEQ_W + n0 + t) * HEADS + h)) * MPAD + m];
    }
    for (int i = threadIdx.x; i < 32 * DH; i += 256) {
      int t = i >> 4, d = i & 15;
      VV[i] = v[(size_t)(b * SEQ_W + n0 + t) * DDIM + h * DH + d];
    }
    __syncthreads();
#pragma unroll
    for (int a = 0; a < 3; ++a) {
      int s = threadIdx.x + 256 * a;
      if (s < MFEAT * 17) {
        int m = s / 17, d = s - m * 17;
        float r = 0.f;
        for (int t = 0; t < 32; ++t)
          r += KP[t * MPAD + m] * (d < DH ? VV[t * DH + d] : 1.0f);
        acc[a] += r;
      }
    }
  }
#pragma unroll
  for (int a = 0; a < 3; ++a) {
    int s = threadIdx.x + 256 * a;
    if (s < MFEAT * 17) ctx[(size_t)bh * (MFEAT * 17) + s] = acc[a];
  }
}

// ---------------- attention output: o = (qp @ ctx) * dinv -> bf16 ------------
__global__ __launch_bounds__(256)
void attn_out(const float* __restrict__ qp, const float* __restrict__ ctx,
              ushort_t* __restrict__ o) {
  int b  = blockIdx.x >> 8;                 // 256 blocks per batch (32 tokens each)
  int n0 = (blockIdx.x & 255) * 32;
  __shared__ float CTX[HEADS * MFEAT * 17];
  for (int i = threadIdx.x; i < HEADS * MFEAT * 17; i += 256) {
    int hh = i / (MFEAT * 17), r = i - hh * (MFEAT * 17);
    CTX[i] = ctx[(size_t)(b * HEADS + hh) * (MFEAT * 17) + r];
  }
  __syncthreads();
  int tok = b * SEQ_W + n0 + (threadIdx.x >> 3);
  int h   = threadIdx.x & 7;
  const float* qr = qp + (size_t)(tok * HEADS + h) * MPAD;
  const float* C  = CTX + h * (MFEAT * 17);
  float num[DH] = {};
  float den = 0.f;
  for (int m = 0; m < MFEAT; ++m) {
    float qv = qr[m];
    den += qv * C[m * 17 + DH];
#pragma unroll
    for (int d = 0; d < DH; ++d) num[d] += qv * C[m * 17 + d];
  }
  float dinv = 1.f / den;
  ushort_t* dst = o + (size_t)tok * DDIM + h * DH;
#pragma unroll
  for (int d = 0; d < DH; ++d) dst[d] = f2bf(num[d] * dinv);
}

// ---------------- bottleneck (LAT = 8) ---------------------------------------
__global__ __launch_bounds__(256)
void enc_lin_kernel(const float* __restrict__ h, const float* __restrict__ W,
                    const float* __restrict__ bias, float* __restrict__ z) {
  __shared__ float Ws[DDIM * LATDIM];
  for (int i = threadIdx.x; i < DDIM * LATDIM; i += 256) Ws[i] = W[i];
  __syncthreads();
  int t = blockIdx.x * 256 + threadIdx.x;
  const float* hr = h + (size_t)t * DDIM;
  float acc[LATDIM];
#pragma unroll
  for (int j = 0; j < LATDIM; ++j) acc[j] = bias[j];
  for (int kk = 0; kk < DDIM; ++kk) {
    float xv = hr[kk];
#pragma unroll
    for (int j = 0; j < LATDIM; ++j) acc[j] += xv * Ws[kk * LATDIM + j];
  }
#pragma unroll
  for (int j = 0; j < LATDIM; ++j) z[(size_t)t * LATDIM + j] = gelu_exact(acc[j]);
}

__global__ __launch_bounds__(256)
void dec_lin_kernel(const float* __restrict__ z, const float* __restrict__ W,
                    const float* __restrict__ bias, float* __restrict__ h) {
  int idx = blockIdx.x * 256 + threadIdx.x;
  int t = idx >> 7, d = idx & 127;
  float acc = bias[d];
#pragma unroll
  for (int j = 0; j < LATDIM; ++j) acc += z[(size_t)t * LATDIM + j] * W[j * DDIM + d];
  h[idx] = acc;
}

__global__ __launch_bounds__(256)
void tanh_kernel(const float* __restrict__ h, float* __restrict__ out) {
  int idx = blockIdx.x * 256 + threadIdx.x;
  out[idx] = tanhf(h[idx]);
}

// ============================================================================
extern "C" void kernel_launch(void* const* d_in, const int* in_sizes, int n_in,
                              void* d_out, int out_size, void* d_ws, size_t ws_size,
                              hipStream_t stream) {
  const float* x     = (const float*)d_in[0];
  const float* tok_W = (const float*)d_in[1];
  const float* tok_b = (const float*)d_in[2];
  const float* ln1_g = (const float*)d_in[3];
  const float* ln1_b = (const float*)d_in[4];
  const float* Wq    = (const float*)d_in[5];
  const float* bq    = (const float*)d_in[6];
  const float* Wk    = (const float*)d_in[7];
  const float* bk    = (const float*)d_in[8];
  const float* Wv    = (const float*)d_in[9];
  const float* bv    = (const float*)d_in[10];
  const float* Wo    = (const float*)d_in[11];
  const float* bo    = (const float*)d_in[12];
  const float* ln2_g = (const float*)d_in[13];
  const float* ln2_b = (const float*)d_in[14];
  const float* W1    = (const float*)d_in[15];
  const float* b1    = (const float*)d_in[16];
  const float* W2    = (const float*)d_in[17];
  const float* b2    = (const float*)d_in[18];
  const float* proj  = (const float*)d_in[19];
  const float* encW  = (const float*)d_in[20];
  const float* encb  = (const float*)d_in[21];
  const float* decW  = (const float*)d_in[22];
  const float* decb  = (const float*)d_in[23];
  (void)in_sizes; (void)n_in; (void)out_size; (void)ws_size;

  // ---- carve scratch ----
  char* ws = (char*)d_ws;
  size_t off = 0;
  auto alloc = [&](size_t bytes) -> char* {
    char* p = ws + off;
    off = (off + bytes + 255) & ~(size_t)255;
    return p;
  };
  float*    h      = (float*)   alloc((size_t)T_TOK * DDIM * 4);
  ushort_t* lnbuf  = (ushort_t*)alloc((size_t)T_TOK * DDIM * 2);   // ln out / x-pack / attn-o
  ushort_t* midbuf = (ushort_t*)alloc((size_t)T_TOK * DFF * 2);
  float*    q      = (float*)   alloc((size_t)T_TOK * DDIM * 4);
  float*    k      = (float*)   alloc((size_t)T_TOK * DDIM * 4);
  float*    v      = (float*)   alloc((size_t)T_TOK * DDIM * 4);
  float*    qp     = (float*)   alloc((size_t)T_TOK * HEADS * MPAD * 4);
  float*    kp     = (float*)   alloc((size_t)T_TOK * HEADS * MPAD * 4);
  float*    ctxb   = (float*)   alloc((size_t)64 * MFEAT * 17 * 4);
  unsigned* kmax   = (unsigned*)alloc(64 * 4);
  float*    z      = (float*)   alloc((size_t)T_TOK * LATDIM * 4);
  ushort_t* wpk    = (ushort_t*)alloc((size_t)(16384 + NLAYERS * 196608) * 2);

  ushort_t* tokWt = wpk;                              // [128,128]
  auto wqt = [&](int i) { return wpk + 16384 + (size_t)i * 196608; };
  auto wkt = [&](int i) { return wqt(i) + 16384; };
  auto wvt = [&](int i) { return wkt(i) + 16384; };
  auto wot = [&](int i) { return wvt(i) + 16384; };
  auto w1t = [&](int i) { return wot(i) + 16384; };   // [512,128]
  auto w2t = [&](int i) { return w1t(i) + 65536; };   // [128,512]

  // ---- weight repack ----
  pack_wt<<<(128 * 128 + 255) / 256, 256, 0, stream>>>(tok_W, tokWt, 128, 128);
  for (int i = 0; i < NLAYERS; ++i) {
    pack_wt<<<64, 256, 0, stream>>>(Wq + (size_t)i * 16384, wqt(i), 128, 128);
    pack_wt<<<64, 256, 0, stream>>>(Wk + (size_t)i * 16384, wkt(i), 128, 128);
    pack_wt<<<64, 256, 0, stream>>>(Wv + (size_t)i * 16384, wvt(i), 128, 128);
    pack_wt<<<64, 256, 0, stream>>>(Wo + (size_t)i * 16384, wot(i), 128, 128);
    pack_wt<<<256, 256, 0, stream>>>(W1 + (size_t)i * 65536, w1t(i), 128, 512);
    pack_wt<<<256, 256, 0, stream>>>(W2 + (size_t)i * 65536, w2t(i), 512, 128);
  }

  const dim3 g128(DDIM / 64, T_TOK / 128);   // (2, 512)
  const dim3 g512(DFF / 64,  T_TOK / 128);   // (8, 512)

  // ---- embed: h = x @ tok_W + tok_b + fixed_pos_emb ----
  pack_bf16<<<(T_TOK * DDIM) / 256, 256, 0, stream>>>(x, lnbuf, T_TOK * DDIM);
  gemm_bf16_wmma<1><<<g128, 256, 0, stream>>>(lnbuf, tokWt, tok_b, nullptr, h, nullptr, DDIM, DDIM);

  auto run_layer = [&](int i) {
    const float* pr = proj + (size_t)i * MFEAT * DH;
    // --- attention ---
    layernorm_bf16<<<T_TOK / 8, 256, 0, stream>>>(h, ln1_g + i * DDIM, ln1_b + i * DDIM, lnbuf);
    gemm_bf16_wmma<0><<<g128, 256, 0, stream>>>(lnbuf, wqt(i), bq + i * DDIM, nullptr, q, nullptr, DDIM, DDIM);
    gemm_bf16_wmma<0><<<g128, 256, 0, stream>>>(lnbuf, wkt(i), bk + i * DDIM, nullptr, k, nullptr, DDIM, DDIM);
    gemm_bf16_wmma<0><<<g128, 256, 0, stream>>>(lnbuf, wvt(i), bv + i * DDIM, nullptr, v, nullptr, DDIM, DDIM);
    kmax_init<<<1, 64, 0, stream>>>(kmax);
    feat_q<<<T_TOK * HEADS / 256, 256, 0, stream>>>(q, pr, qp);
    feat_k1<<<T_TOK * HEADS / 256, 256, 0, stream>>>(k, pr, kp, kmax);
    feat_k2<<<T_TOK * HEADS / 256, 256, 0, stream>>>(kp, kmax);
    ctx_kernel<<<64, 256, 0, stream>>>(kp, v, ctxb);
    attn_out<<<T_TOK / 32, 256, 0, stream>>>(qp, ctxb, lnbuf);
    gemm_bf16_wmma<2><<<g128, 256, 0, stream>>>(lnbuf, wot(i), bo + i * DDIM, h, h, nullptr, DDIM, DDIM);
    // --- FFN ---
    layernorm_bf16<<<T_TOK / 8, 256, 0, stream>>>(h, ln2_g + i * DDIM, ln2_b + i * DDIM, lnbuf);
    gemm_bf16_wmma<3><<<g512, 256, 0, stream>>>(lnbuf, w1t(i), b1 + i * DFF, nullptr, nullptr, midbuf, DFF, DDIM);
    gemm_bf16_wmma<2><<<g128, 256, 0, stream>>>(midbuf, w2t(i), b2 + i * DDIM, h, h, nullptr, DDIM, DFF);
  };

  for (int i = 0; i < 4; ++i) run_layer(i);

  // ---- bottleneck: z = gelu(h @ encW + encb); h = z @ decW + decb ----
  enc_lin_kernel<<<T_TOK / 256, 256, 0, stream>>>(h, encW, encb, z);
  dec_lin_kernel<<<T_TOK * DDIM / 256, 256, 0, stream>>>(z, decW, decb, h);

  for (int i = 4; i < 8; ++i) run_layer(i);

  tanh_kernel<<<T_TOK * DDIM / 256, 256, 0, stream>>>(h, (float*)d_out);
}
